// GCN_44744969290503
// MI455X (gfx1250) — matrix operand order repaired and checked by reading, not compiled, not run on hardware
//
#include <hip/hip_runtime.h>
#include <hip/hip_bf16.h>

typedef __attribute__((ext_vector_type(16))) _Float16 v16h;
typedef __attribute__((ext_vector_type(8)))  _Float16 v8h;
typedef __attribute__((ext_vector_type(8)))  float    v8f;

// ---------------------------------------------------------------------------
// Elementwise helpers
// ---------------------------------------------------------------------------
__global__ void gcn_cast_f16(const float* __restrict__ in, _Float16* __restrict__ out, int n) {
  int t = blockIdx.x * blockDim.x + threadIdx.x;
  if (t < n) out[t] = (_Float16)in[t];
}

__global__ void gcn_relu_cast_f16(const float* __restrict__ in, _Float16* __restrict__ out, int n) {
  int t = blockIdx.x * blockDim.x + threadIdx.x;
  if (t < n) out[t] = (_Float16)fmaxf(in[t], 0.0f);
}

__global__ void gcn_fill(float* __restrict__ p, float v, int n) {
  int t = blockIdx.x * blockDim.x + threadIdx.x;
  if (t < n) p[t] = v;
}

__global__ void gcn_rsqrt_inplace(float* __restrict__ p, int n) {
  int t = blockIdx.x * blockDim.x + threadIdx.x;
  if (t < n) p[t] = rsqrtf(p[t]);
}

// deg[dst[e]] += 1  (self-loop pre-seeded with fill=1.0)
__global__ void gcn_deg_edges(const int* __restrict__ dst, float* __restrict__ deg, int e) {
  int t = blockIdx.x * blockDim.x + threadIdx.x;
  if (t < e) atomicAdd(&deg[dst[t]], 1.0f);
}

// ---------------------------------------------------------------------------
// Pack weight matrix W[K][Nc] (f32, row-major) into the CDNA5 WMMA B-operand
// register layout for V_WMMA_F32_16X16X32_F16:
//   B is 32(K) x 16(N) per tile; lane = N + 16*h; slot i<8 -> K = 8h + i, i>=8 -> K = 16 + 8h + (i-8)
// Packed index: ((kt*NT + nt)*32 + lane)*16 + slot
// ---------------------------------------------------------------------------
__global__ void gcn_pack_b(const float* __restrict__ W, _Float16* __restrict__ out, int K, int Nc) {
  int t = blockIdx.x * blockDim.x + threadIdx.x;
  if (t >= K * Nc) return;
  int slot = t & 15;
  int lane = (t >> 4) & 31;
  int rest = t >> 9;
  int NT   = Nc >> 4;
  int nt   = rest % NT;
  int kt   = rest / NT;
  int k = kt * 32 + ((lane & 16) ? 8 : 0) + (slot < 8 ? slot : slot + 8);
  int n = nt * 16 + (lane & 15);
  out[t] = (_Float16)W[(size_t)k * Nc + n];
}

// ---------------------------------------------------------------------------
// WMMA GEMM: C[M,Nc] = A[M,K](f16) * Bpacked (f16), f32 accumulate.
// One wave (32 threads) per 16x16 tile. grid = (M/16, Nc/16).
// A-fragment (16-bit A 16x32 layout): lane half -> K-base 0/8; slots 0..7 are
// K=base..base+7 (one 16B load), slots 8..15 are K=16+base..16+base+7.
// fuse!=0: C = relu(A*B + bias[col]) (MLP head); else plain A*B.
// ---------------------------------------------------------------------------
__global__ void gcn_wmma_gemm(const _Float16* __restrict__ A, const _Float16* __restrict__ Bp,
                              const float* __restrict__ bias, float* __restrict__ C,
                              int K, int Nc, int fuse) {
  const int mt   = blockIdx.x;
  const int nt   = blockIdx.y;
  const int lane = threadIdx.x;          // 0..31, one wave
  const int NT   = Nc >> 4;
  const int KT   = K >> 5;

  v8f acc = {};
  const _Float16* arow = A + (size_t)(mt * 16 + (lane & 15)) * (size_t)K + ((lane & 16) ? 8 : 0);
  const _Float16* bp   = Bp + (size_t)nt * 512 + (size_t)lane * 16;

  for (int kt = 0; kt < KT; ++kt) {
    v8h lo = *(const v8h*)(arow + kt * 32);
    v8h hi = *(const v8h*)(arow + kt * 32 + 16);
    v16h a;
#pragma unroll
    for (int i = 0; i < 8; ++i) { a[i] = lo[i]; a[i + 8] = hi[i]; }
    v16h b = *(const v16h*)(bp + (size_t)kt * NT * 512);
    acc = __builtin_amdgcn_wmma_f32_16x16x32_f16(
        /*neg_a=*/false, a, /*neg_b=*/false, b,
        /*c_mod=*/(short)0, acc, /*reuse_a=*/false, /*reuse_b=*/false);
  }

  const int col  = nt * 16 + (lane & 15);
  const int row0 = mt * 16 + ((lane & 16) ? 8 : 0);
  const float badd = fuse ? bias[col] : 0.0f;
#pragma unroll
  for (int j = 0; j < 8; ++j) {
    float v = acc[j] + badd;
    if (fuse) v = fmaxf(v, 0.0f);
    C[(size_t)(row0 + j) * Nc + col] = v;
  }
}

// agg[i][c] = hg[i][c]*dinv[i]^2 + bias[c]   (self-loop term + conv bias)
__global__ void gcn_selfloop_bias(const float* __restrict__ hg, const float* __restrict__ dinv,
                                  const float* __restrict__ bias, float* __restrict__ agg, int n) {
  int t = blockIdx.x * blockDim.x + threadIdx.x;
  if (t >= n) return;
  int i = t >> 6, c = t & 63;
  float di = dinv[i];
  agg[t] = hg[t] * di * di + bias[c];
}

// agg[dst[e]][c] += hg[src[e]][c] * dinv[src]*dinv[dst]; 2 channels per thread
__global__ void gcn_edge_scatter(const float* __restrict__ hg, const float* __restrict__ dinv,
                                 const int* __restrict__ src, const int* __restrict__ dst,
                                 float* __restrict__ agg, int e) {
  int t = blockIdx.x * blockDim.x + threadIdx.x;
  if (t >= e * 32) return;
  int ei = t >> 5;
  int c0 = (t & 31) * 2;
  int s = src[ei], d = dst[ei];
  float w = dinv[s] * dinv[d];
  float2 hv = *(const float2*)(hg + (size_t)s * 64 + c0);
  atomicAdd(agg + (size_t)d * 64 + c0,     hv.x * w);
  atomicAdd(agg + (size_t)d * 64 + c0 + 1, hv.y * w);
}

// g[batch[i]][c] += agg[i][c]; 2 channels per thread
__global__ void gcn_pool(const float* __restrict__ agg, const int* __restrict__ batch,
                         float* __restrict__ g, int n) {
  int t = blockIdx.x * blockDim.x + threadIdx.x;
  if (t >= n * 32) return;
  int i = t >> 5;
  int c0 = (t & 31) * 2;
  int b = batch[i];
  float2 hv = *(const float2*)(agg + (size_t)i * 64 + c0);
  atomicAdd(g + (size_t)b * 64 + c0,     hv.x);
  atomicAdd(g + (size_t)b * 64 + c0 + 1, hv.y);
}

// out[gi][o] = g1[gi] . lin2_w[:,o] + lin2_b[o]   (OUT_CH=10, not WMMA-friendly)
__global__ void gcn_lin2(const float* __restrict__ g1, const float* __restrict__ W,
                         const float* __restrict__ b, float* __restrict__ out, int G) {
  int t = blockIdx.x * blockDim.x + threadIdx.x;
  if (t >= G * 10) return;
  int gi = t / 10, o = t % 10;
  float s = b[o];
  const float* gp = g1 + (size_t)gi * 128;
#pragma unroll 8
  for (int k = 0; k < 128; ++k) s += gp[k] * W[k * 10 + o];
  out[t] = s;
}

// ---------------------------------------------------------------------------
extern "C" void kernel_launch(void* const* d_in, const int* in_sizes, int n_in,
                              void* d_out, int out_size, void* d_ws, size_t ws_size,
                              hipStream_t stream) {
  const float* x    = (const float*)d_in[0];
  const int*   ei   = (const int*)d_in[1];
  const int*   batch= (const int*)d_in[2];
  const float* W0   = (const float*)d_in[3];  const float* b0 = (const float*)d_in[4];
  const float* W1   = (const float*)d_in[5];  const float* b1 = (const float*)d_in[6];
  const float* W2   = (const float*)d_in[7];  const float* b2 = (const float*)d_in[8];
  const float* L1w  = (const float*)d_in[9];  const float* L1b = (const float*)d_in[10];
  const float* L2w  = (const float*)d_in[11]; const float* L2b = (const float*)d_in[12];
  float* out = (float*)d_out;

  const int N = in_sizes[0] / 128;   // 50000 (multiple of 16)
  const int E = in_sizes[1] / 2;     // 800000
  const int G = out_size / 10;       // 512
  const int* src = ei;               // edge_index[0]
  const int* dst = ei + E;           // edge_index[1]

  // Carve workspace (256B-aligned slices)
  char* w = (char*)d_ws;
  auto carve = [&](size_t bytes) -> char* {
    char* p = w; w += (bytes + 255) & ~(size_t)255; return p;
  };
  float*     dinv = (float*)    carve((size_t)N * 4);
  _Float16*  x16  = (_Float16*) carve((size_t)N * 128 * 2);
  _Float16*  h16  = (_Float16*) carve((size_t)N * 64 * 2);
  float*     hg   = (float*)    carve((size_t)N * 64 * 4);
  float*     agg  = (float*)    carve((size_t)N * 64 * 4);
  _Float16*  W0p  = (_Float16*) carve(128 * 64 * 2);
  _Float16*  W1p  = (_Float16*) carve(64 * 64 * 2);
  _Float16*  W2p  = (_Float16*) carve(64 * 64 * 2);
  _Float16*  L1p  = (_Float16*) carve(64 * 128 * 2);
  float*     g    = (float*)    carve((size_t)G * 64 * 4);
  _Float16*  g16  = (_Float16*) carve((size_t)G * 64 * 2);
  float*     g1   = (float*)    carve((size_t)G * 128 * 4);

  const int B = 256;
  auto nb = [&](int n) { return (n + B - 1) / B; };

  // Prep: cast features, pack weights into WMMA-B layout
  gcn_cast_f16<<<nb(N * 128), B, 0, stream>>>(x, x16, N * 128);
  gcn_pack_b<<<nb(128 * 64), B, 0, stream>>>(W0, W0p, 128, 64);
  gcn_pack_b<<<nb(64 * 64),  B, 0, stream>>>(W1, W1p, 64, 64);
  gcn_pack_b<<<nb(64 * 64),  B, 0, stream>>>(W2, W2p, 64, 64);
  gcn_pack_b<<<nb(64 * 128), B, 0, stream>>>(L1w, L1p, 64, 128);

  // Degree (incl. self-loop) -> dinv = rsqrt(deg)
  gcn_fill<<<nb(N), B, 0, stream>>>(dinv, 1.0f, N);
  gcn_deg_edges<<<nb(E), B, 0, stream>>>(dst, dinv, E);
  gcn_rsqrt_inplace<<<nb(N), B, 0, stream>>>(dinv, N);

  dim3 gconv(N / 16, 4);   // 3125 x 4 tiles, one wave each

  // Layer 0: h = relu(scatter(x@W0) + b0)
  gcn_wmma_gemm<<<gconv, 32, 0, stream>>>(x16, W0p, nullptr, hg, 128, 64, 0);
  gcn_selfloop_bias<<<nb(N * 64), B, 0, stream>>>(hg, dinv, b0, agg, N * 64);
  gcn_edge_scatter<<<nb(E * 32), B, 0, stream>>>(hg, dinv, src, dst, agg, E);
  gcn_relu_cast_f16<<<nb(N * 64), B, 0, stream>>>(agg, h16, N * 64);

  // Layer 1
  gcn_wmma_gemm<<<gconv, 32, 0, stream>>>(h16, W1p, nullptr, hg, 64, 64, 0);
  gcn_selfloop_bias<<<nb(N * 64), B, 0, stream>>>(hg, dinv, b1, agg, N * 64);
  gcn_edge_scatter<<<nb(E * 32), B, 0, stream>>>(hg, dinv, src, dst, agg, E);
  gcn_relu_cast_f16<<<nb(N * 64), B, 0, stream>>>(agg, h16, N * 64);

  // Layer 2 (no relu)
  gcn_wmma_gemm<<<gconv, 32, 0, stream>>>(h16, W2p, nullptr, hg, 64, 64, 0);
  gcn_selfloop_bias<<<nb(N * 64), B, 0, stream>>>(hg, dinv, b2, agg, N * 64);
  gcn_edge_scatter<<<nb(E * 32), B, 0, stream>>>(hg, dinv, src, dst, agg, E);

  // Graph sum-pool
  gcn_fill<<<nb(G * 64), B, 0, stream>>>(g, 0.0f, G * 64);
  gcn_pool<<<nb(N * 32), B, 0, stream>>>(agg, batch, g, N);

  // MLP head: relu(g@lin1 + b) via WMMA (fused epilogue), then scalar lin2
  gcn_cast_f16<<<nb(G * 64), B, 0, stream>>>(g, g16, G * 64);
  dim3 glin(G / 16, 8);    // 32 x 8 tiles
  gcn_wmma_gemm<<<glin, 32, 0, stream>>>(g16, L1p, L1b, g1, 64, 128, 1);
  gcn_lin2<<<nb(G * 10), B, 0, stream>>>(g1, L2w, L2b, out, G);
}
